// RSSM_73014444032776
// MI455X (gfx1250) — matrix operand order repaired and checked by reading, not compiled, not run on hardware
//
#include <hip/hip_runtime.h>
#include <hip/hip_bf16.h>
#include <math.h>

// ---------------------------------------------------------------------------
// RSSM rollout for MI455X (gfx1250): bf16 WMMA GEMMs with fused epilogues.
// GRU GEMM uses async global->LDS staging (ASYNCcnt) with double buffering.
// ---------------------------------------------------------------------------

typedef __bf16 bf16;
typedef __attribute__((ext_vector_type(16))) __bf16 v16bf;
typedef __attribute__((ext_vector_type(8)))  __bf16 v8bf;
typedef __attribute__((ext_vector_type(8)))  float  v8f;

#define T_STEPS 50
#define BATCH   256
#define DETER   1024
#define STOCH   64
#define NODE    512
#define EMB     1024
#define ACT     32
#define SAK     96                      // STOCH + ACT
#define OUTC    (6*STOCH + 2*DETER)     // 2432 floats per (t,b)
#define KC      64                      // GRU K-chunk staged in LDS

#define WMMA_BF16(a,b,c) \
  __builtin_amdgcn_wmma_f32_16x16x32_bf16(false,(a),false,(b),(short)0,(c),false,false)

// ---- fragment loaders (wave32; contiguous k order shared by A and B) --------------

// A-matrix 16x32 bf16: lane m in [0,16) holds row m; elements 0..15 -> K = ks..ks+15,
// ks = (lane>=16) ? 16 : 0 (contiguous, matches B).
__device__ __forceinline__ v16bf load_fragA(const bf16* __restrict__ p, int ld,
                                            int r0, int k0, int lane) {
  const bf16* q = p + (size_t)(r0 + (lane & 15)) * ld + k0 + ((lane >> 4) << 4);
  v8bf lo = *(const v8bf*)(q);
  v8bf hi = *(const v8bf*)(q + 8);
  v16bf f;
#pragma unroll
  for (int i = 0; i < 8; ++i) { f[i] = lo[i]; f[i + 8] = hi[i]; }
  return f;
}

// B-matrix 32x16 bf16: lane n in [0,16) holds column n; lanes 0-15 hold K=0..15,
// lanes 16-31 hold K=16..31 (contiguous). Weight stored [out, in] row-major,
// B[k][n] = W[n][k] -> one contiguous 16-element run of row n.
__device__ __forceinline__ v16bf load_fragB(const bf16* __restrict__ p, int ld,
                                            int n0, int k0, int lane) {
  const bf16* q = p + (size_t)(n0 + (lane & 15)) * ld + k0 + ((lane >> 4) << 4);
  v8bf lo = *(const v8bf*)(q);
  v8bf hi = *(const v8bf*)(q + 8);
  v16bf f;
#pragma unroll
  for (int i = 0; i < 8; ++i) { f[i] = lo[i]; f[i + 8] = hi[i]; }
  return f;
}

__device__ __forceinline__ float sigmoidf_(float x) { return 1.f / (1.f + expf(-x)); }
__device__ __forceinline__ float eluf_(float x)     { return x > 0.f ? x : (expf(x) - 1.f); }
__device__ __forceinline__ float softplusf_(float x){ return x > 20.f ? x : log1pf(expf(x)); }

// gfx1250 async global->LDS copy (16B per lane), tracked by ASYNCcnt.
__device__ __forceinline__ void async_copy_b128(unsigned lds_off, const bf16* gaddr) {
  asm volatile("global_load_async_to_lds_b128 %0, %1, off"
               :: "v"(lds_off), "v"((unsigned long long)(uintptr_t)gaddr)
               : "memory");
}
__device__ __forceinline__ void wait_async0() {
  asm volatile("s_wait_asynccnt 0x0" ::: "memory");
}

// ---- preprocessing --------------------------------------------------------

__global__ void k_cvt(const float* __restrict__ src, bf16* __restrict__ dst, int n) {
  int i = blockIdx.x * blockDim.x + threadIdx.x;
  if (i < n) dst[i] = (bf16)src[i];
}

__global__ void k_copy(const float* __restrict__ src, float* __restrict__ dst, int n) {
  int i = blockIdx.x * blockDim.x + threadIdx.x;
  if (i < n) dst[i] = src[i];
}

// Per-step input prep: h = deter*nt (f32 + bf16), sa = [stoch*nt, action*nt] (bf16)
__global__ void k_prep(const float* __restrict__ action, const float* __restrict__ nonterms,
                       const float* __restrict__ stoch_f, const float* __restrict__ deter_f,
                       bf16* __restrict__ sa_bf, float* __restrict__ h_f,
                       bf16* __restrict__ h_bf, int t) {
  int idx = blockIdx.x * blockDim.x + threadIdx.x;
  const int NH = BATCH * DETER;
  if (idx < NH) {
    int b = idx / DETER;
    float ntv = nonterms[t * BATCH + b];
    float hv = deter_f[idx] * ntv;
    h_f[idx]  = hv;
    h_bf[idx] = (bf16)hv;
  } else {
    int i2 = idx - NH;
    if (i2 < BATCH * SAK) {
      int b = i2 / SAK, c = i2 % SAK;
      float ntv = nonterms[t * BATCH + b];
      float v = (c < STOCH) ? stoch_f[b * STOCH + c]
                            : action[((size_t)t * BATCH + b) * ACT + (c - STOCH)];
      sa_bf[i2] = (bf16)(v * ntv);
    }
  }
}

// ---- GEMM 1: embed = elu(sa @ W_embed^T + b), M=256 N=1024 K=96 -----------
__global__ void __launch_bounds__(128)
k_embed(const bf16* __restrict__ sa, const bf16* __restrict__ Wb,
        const float* __restrict__ bias, bf16* __restrict__ embed_bf,
        float* __restrict__ outp) {
  int lane = threadIdx.x & 31, wave = threadIdx.x >> 5;
  int m0 = blockIdx.x * 16;
  int n0 = blockIdx.y * 256 + wave * 64;
  v8f acc[4] = {};
#pragma unroll
  for (int k0 = 0; k0 < SAK; k0 += 32) {
    v16bf a = load_fragA(sa, SAK, m0, k0, lane);
#pragma unroll
    for (int j = 0; j < 4; ++j) {
      v16bf bm = load_fragB(Wb, SAK, n0 + j * 16, k0, lane);
      acc[j] = WMMA_BF16(a, bm, acc[j]);
    }
  }
  int n = lane & 15, mb = (lane >> 4) << 3;
#pragma unroll
  for (int j = 0; j < 4; ++j) {
    int col = n0 + j * 16 + n;
    float bb = bias[col];
#pragma unroll
    for (int v = 0; v < 8; ++v) {
      int m = m0 + mb + v;
      float e = eluf_(acc[j][v] + bb);
      embed_bf[(size_t)m * EMB + col] = (bf16)e;
      outp[(size_t)m * OUTC + 6 * STOCH + DETER + col] = e;
    }
  }
}

// ---- GEMM 2: GRU cell. 8 waves/block, 32x64 tile, async-LDS staged A ------
__global__ void __launch_bounds__(256)
k_gru(const bf16* __restrict__ embed, const bf16* __restrict__ hbf,
      const float* __restrict__ hf, const bf16* __restrict__ Wih,
      const bf16* __restrict__ Whh, const float* __restrict__ bih,
      const float* __restrict__ bhh, float* __restrict__ deter_f,
      bf16* __restrict__ deter_bf, float* __restrict__ outp) {
  // double-buffered staging: [buf][mat(embed/h)][32 rows][KC]
  __shared__ bf16 smem[2][2][32][KC];

  int tid  = threadIdx.x;
  int lane = tid & 31, wave = tid >> 5;
  int mi = wave >> 2, ni = wave & 3;       // 2 M-tiles x 4 N-tiles
  int mBlk = blockIdx.x * 32;
  int m0 = mBlk + mi * 16;
  int j0 = blockIdx.y * 64 + ni * 16;

  unsigned lds0 = (unsigned)(uintptr_t)(&smem[0][0][0][0]);
  int row = tid >> 3, seg = tid & 7;       // 256 threads cover 2x32x(KC=64)

  v8f a_ir = {}, a_iz = {}, a_in = {}, a_hr = {}, a_hz = {}, a_hn = {};

  const int NC = DETER / KC;
  // issue chunk 0
  {
    const bf16* ge = embed + (size_t)(mBlk + row) * DETER + seg * 8;
    const bf16* gh = hbf   + (size_t)(mBlk + row) * DETER + seg * 8;
    async_copy_b128(lds0 + (unsigned)(((0 * 32 + row) * KC + seg * 8) * 2), ge);
    async_copy_b128(lds0 + (unsigned)(((1 * 32 + row) * KC + seg * 8) * 2), gh);
  }
  for (int c = 0; c < NC; ++c) {
    int buf = c & 1;
    wait_async0();
    __syncthreads();                       // staged chunk c visible to all waves
    if (c + 1 < NC) {                      // prefetch chunk c+1 into other buffer
      int kc = (c + 1) * KC;
      const bf16* ge = embed + (size_t)(mBlk + row) * DETER + kc + seg * 8;
      const bf16* gh = hbf   + (size_t)(mBlk + row) * DETER + kc + seg * 8;
      unsigned b2 = (unsigned)((c + 1) & 1) * 2u;
      async_copy_b128(lds0 + (unsigned)((((b2 + 0) * 32 + row) * KC + seg * 8) * 2), ge);
      async_copy_b128(lds0 + (unsigned)((((b2 + 1) * 32 + row) * KC + seg * 8) * 2), gh);
    }
    const bf16* tE = &smem[buf][0][0][0];
    const bf16* tH = &smem[buf][1][0][0];
#pragma unroll
    for (int kk = 0; kk < KC; kk += 32) {
      int k0 = c * KC + kk;
      v16bf ae = load_fragA(tE, KC, mi * 16, kk, lane);
      v16bf ah = load_fragA(tH, KC, mi * 16, kk, lane);
      v16bf b0 = load_fragB(Wih, DETER, j0,             k0, lane);
      a_ir = WMMA_BF16(ae, b0, a_ir);
      v16bf b1 = load_fragB(Wih, DETER, DETER + j0,     k0, lane);
      a_iz = WMMA_BF16(ae, b1, a_iz);
      v16bf b2 = load_fragB(Wih, DETER, 2 * DETER + j0, k0, lane);
      a_in = WMMA_BF16(ae, b2, a_in);
      v16bf b3 = load_fragB(Whh, DETER, j0,             k0, lane);
      a_hr = WMMA_BF16(ah, b3, a_hr);
      v16bf b4 = load_fragB(Whh, DETER, DETER + j0,     k0, lane);
      a_hz = WMMA_BF16(ah, b4, a_hz);
      v16bf b5 = load_fragB(Whh, DETER, 2 * DETER + j0, k0, lane);
      a_hn = WMMA_BF16(ah, b5, a_hn);
    }
    __syncthreads();                       // all waves done reading buf before reuse
  }

  int n = lane & 15, mb = (lane >> 4) << 3;
  int j = j0 + n;
  float br_i = bih[j], bz_i = bih[DETER + j], bn_i = bih[2 * DETER + j];
  float br_h = bhh[j], bz_h = bhh[DETER + j], bn_h = bhh[2 * DETER + j];
#pragma unroll
  for (int v = 0; v < 8; ++v) {
    int m = m0 + mb + v;
    float r  = sigmoidf_((a_ir[v] + br_i) + (a_hr[v] + br_h));
    float z  = sigmoidf_((a_iz[v] + bz_i) + (a_hz[v] + bz_h));
    float nn = tanhf((a_in[v] + bn_i) + r * (a_hn[v] + bn_h));
    float h  = hf[(size_t)m * DETER + j];
    float d  = (1.f - z) * nn + z * h;
    deter_f[(size_t)m * DETER + j]  = d;
    deter_bf[(size_t)m * DETER + j] = (bf16)d;
    outp[(size_t)m * OUTC + 6 * STOCH + j] = d;
  }
}

// ---- GEMM 3: prior/posterior hidden layers (z = 0 prior K=1024, z = 1 post K=2048)
__global__ void __launch_bounds__(128)
k_hidden(const bf16* __restrict__ deter_bf, const bf16* __restrict__ obs_bf_t,
         const bf16* __restrict__ Wp1, const float* __restrict__ bp1,
         const bf16* __restrict__ Wq1, const float* __restrict__ bq1,
         bf16* __restrict__ h1, bf16* __restrict__ hq) {
  int lane = threadIdx.x & 31, wave = threadIdx.x >> 5;
  int z = blockIdx.z;
  int m0 = blockIdx.x * 16;
  int n0 = blockIdx.y * 256 + wave * 64;
  const int K = z ? (DETER + EMB) : DETER;
  const bf16*  Wmat = z ? Wq1 : Wp1;
  const float* bias = z ? bq1 : bp1;
  bf16* dst = z ? hq : h1;
  v8f acc[4] = {};
  for (int k0 = 0; k0 < K; k0 += 32) {
    v16bf a = (k0 < DETER) ? load_fragA(deter_bf, DETER, m0, k0, lane)
                           : load_fragA(obs_bf_t, EMB, m0, k0 - DETER, lane);
#pragma unroll
    for (int j = 0; j < 4; ++j) {
      v16bf bm = load_fragB(Wmat, K, n0 + j * 16, k0, lane);
      acc[j] = WMMA_BF16(a, bm, acc[j]);
    }
  }
  int n = lane & 15, mb = (lane >> 4) << 3;
#pragma unroll
  for (int j = 0; j < 4; ++j) {
    int col = n0 + j * 16 + n;
    float bb = bias[col];
#pragma unroll
    for (int v = 0; v < 8; ++v) {
      int m = m0 + mb + v;
      dst[(size_t)m * NODE + col] = (bf16)eluf_(acc[j][v] + bb);
    }
  }
}

// ---- GEMM 4: heads, fused mean/std/sample (z = 0 prior, z = 1 posterior) --
__global__ void __launch_bounds__(128)
k_head(const bf16* __restrict__ h1, const bf16* __restrict__ hq,
       const bf16* __restrict__ Wp2, const float* __restrict__ bp2,
       const bf16* __restrict__ Wq2, const float* __restrict__ bq2,
       const float* __restrict__ noise_p_t, const float* __restrict__ noise_q_t,
       float* __restrict__ stoch_f, float* __restrict__ outp) {
  int lane = threadIdx.x & 31, wave = threadIdx.x >> 5;
  int z = blockIdx.z;
  int m0 = blockIdx.x * 16;
  int n0 = wave * 16;                       // 4 waves cover all 64 mean columns
  const bf16*  H   = z ? hq : h1;
  const bf16*  W2  = z ? Wq2 : Wp2;
  const float* b2  = z ? bq2 : bp2;
  const float* eps = z ? noise_q_t : noise_p_t;
  int obase = z ? 3 * STOCH : 0;
  v8f am = {}, ar = {};
  for (int k0 = 0; k0 < NODE; k0 += 32) {
    v16bf a  = load_fragA(H, NODE, m0, k0, lane);
    v16bf bm = load_fragB(W2, NODE, n0,         k0, lane);
    am = WMMA_BF16(a, bm, am);
    v16bf br = load_fragB(W2, NODE, STOCH + n0, k0, lane);
    ar = WMMA_BF16(a, br, ar);
  }
  int n = lane & 15, mb = (lane >> 4) << 3;
  int col = n0 + n;
  float bmean = b2[col], braw = b2[STOCH + col];
#pragma unroll
  for (int v = 0; v < 8; ++v) {
    int m = m0 + mb + v;
    float mean = am[v] + bmean;
    float sd   = softplusf_(ar[v] + braw) + 0.1f;
    float st   = mean + sd * eps[(size_t)m * STOCH + col];
    float* ob = outp + (size_t)m * OUTC + obase;
    ob[col]             = mean;
    ob[STOCH + col]     = sd;
    ob[2 * STOCH + col] = st;
    if (z) stoch_f[(size_t)m * STOCH + col] = st;   // carry to next step
  }
}

// ---------------------------------------------------------------------------
extern "C" void kernel_launch(void* const* d_in, const int* in_sizes, int n_in,
                              void* d_out, int out_size, void* d_ws, size_t ws_size,
                              hipStream_t stream) {
  const float* obs      = (const float*)d_in[0];
  const float* action   = (const float*)d_in[1];
  const float* nonterms = (const float*)d_in[2];
  const float* pstoch   = (const float*)d_in[3];
  const float* pdeter   = (const float*)d_in[4];
  const float* noise_p  = (const float*)d_in[5];
  const float* noise_q  = (const float*)d_in[6];
  const float* W_embed  = (const float*)d_in[7];
  const float* b_embed  = (const float*)d_in[8];
  const float* W_ih     = (const float*)d_in[9];
  const float* b_ih     = (const float*)d_in[10];
  const float* W_hh     = (const float*)d_in[11];
  const float* b_hh     = (const float*)d_in[12];
  const float* W_p1     = (const float*)d_in[13];
  const float* b_p1     = (const float*)d_in[14];
  const float* W_p2     = (const float*)d_in[15];
  const float* b_p2     = (const float*)d_in[16];
  const float* W_q1     = (const float*)d_in[17];
  const float* b_q1     = (const float*)d_in[18];
  const float* W_q2     = (const float*)d_in[19];
  const float* b_q2     = (const float*)d_in[20];
  float* out = (float*)d_out;

  char* w = (char*)d_ws;
  size_t off = 0;
  auto alloc = [&](size_t bytes) -> void* {
    void* p = w + off;
    off = (off + bytes + 255) & ~(size_t)255;
    return p;
  };
  bf16* obs_bf  = (bf16*)alloc((size_t)T_STEPS * BATCH * EMB * 2);
  bf16* Wemb_bf = (bf16*)alloc((size_t)DETER * SAK * 2);
  bf16* Wih_bf  = (bf16*)alloc((size_t)3 * DETER * DETER * 2);
  bf16* Whh_bf  = (bf16*)alloc((size_t)3 * DETER * DETER * 2);
  bf16* Wp1_bf  = (bf16*)alloc((size_t)NODE * DETER * 2);
  bf16* Wp2_bf  = (bf16*)alloc((size_t)2 * STOCH * NODE * 2);
  bf16* Wq1_bf  = (bf16*)alloc((size_t)NODE * (DETER + EMB) * 2);
  bf16* Wq2_bf  = (bf16*)alloc((size_t)2 * STOCH * NODE * 2);
  bf16* sa_bf   = (bf16*)alloc((size_t)BATCH * SAK * 2);
  bf16* h_bf    = (bf16*)alloc((size_t)BATCH * DETER * 2);
  bf16* emb_bf  = (bf16*)alloc((size_t)BATCH * EMB * 2);
  bf16* det_bf  = (bf16*)alloc((size_t)BATCH * DETER * 2);
  bf16* h1_bf   = (bf16*)alloc((size_t)BATCH * NODE * 2);
  bf16* hq_bf   = (bf16*)alloc((size_t)BATCH * NODE * 2);
  float* h_f    = (float*)alloc((size_t)BATCH * DETER * 4);
  float* det_f  = (float*)alloc((size_t)BATCH * DETER * 4);
  float* sto_f  = (float*)alloc((size_t)BATCH * STOCH * 4);

  auto cvt = [&](const float* s, bf16* d, size_t n) {
    k_cvt<<<(unsigned)((n + 255) / 256), 256, 0, stream>>>(s, d, (int)n);
  };
  // One-time (per call) conversions to bf16
  cvt(obs,     obs_bf,  (size_t)T_STEPS * BATCH * EMB);
  cvt(W_embed, Wemb_bf, (size_t)DETER * SAK);
  cvt(W_ih,    Wih_bf,  (size_t)3 * DETER * DETER);
  cvt(W_hh,    Whh_bf,  (size_t)3 * DETER * DETER);
  cvt(W_p1,    Wp1_bf,  (size_t)NODE * DETER);
  cvt(W_p2,    Wp2_bf,  (size_t)2 * STOCH * NODE);
  cvt(W_q1,    Wq1_bf,  (size_t)NODE * (DETER + EMB));
  cvt(W_q2,    Wq2_bf,  (size_t)2 * STOCH * NODE);
  k_copy<<<(BATCH * STOCH + 255) / 256, 256, 0, stream>>>(pstoch, sto_f, BATCH * STOCH);
  k_copy<<<(BATCH * DETER + 255) / 256, 256, 0, stream>>>(pdeter, det_f, BATCH * DETER);

  const int prepN = BATCH * DETER + BATCH * SAK;
  for (int t = 0; t < T_STEPS; ++t) {
    float* out_t = out + (size_t)t * BATCH * OUTC;
    const bf16* obs_t = obs_bf + (size_t)t * BATCH * EMB;
    const float* np_t = noise_p + (size_t)t * BATCH * STOCH;
    const float* nq_t = noise_q + (size_t)t * BATCH * STOCH;

    k_prep<<<(prepN + 255) / 256, 256, 0, stream>>>(
        action, nonterms, sto_f, det_f, sa_bf, h_f, h_bf, t);

    k_embed<<<dim3(BATCH / 16, DETER / 256), 128, 0, stream>>>(
        sa_bf, Wemb_bf, b_embed, emb_bf, out_t);

    k_gru<<<dim3(BATCH / 32, DETER / 64), 256, 0, stream>>>(
        emb_bf, h_bf, h_f, Wih_bf, Whh_bf, b_ih, b_hh, det_f, det_bf, out_t);

    k_hidden<<<dim3(BATCH / 16, NODE / 256, 2), 128, 0, stream>>>(
        det_bf, obs_t, Wp1_bf, b_p1, Wq1_bf, b_q1, h1_bf, hq_bf);

    k_head<<<dim3(BATCH / 16, 1, 2), 128, 0, stream>>>(
        h1_bf, hq_bf, Wp2_bf, b_p2, Wq2_bf, b_q2, np_t, nq_t, sto_f, out_t);
  }
}
